// DiTBlock_79611513798969
// MI455X (gfx1250) — matrix operand order repaired and checked by reading, not compiled, not run on hardware
//
#include <hip/hip_runtime.h>
#include <math.h>

// ---------------------------------------------------------------------------
// DiT block for MI455X (gfx1250, wave32, WMMA).
// All GEMMs + both attention einsums run on v_wmma_f32_16x16x32_bf16.
// fp32 inputs are converted once to bf16 in workspace; accumulation is f32.
// Epilogues (bias / residual / GELU / adaLN gating) are fused into the GEMM.
// Row-major global->LDS staging (GEMM A-tile, attention K-tile) uses the
// CDNA5 async copy path (global_load_async_to_lds_b128 + s_wait_asynccnt).
// Workspace use: ~240 MB (weights bf16 + activations).
// ---------------------------------------------------------------------------

#define C_DIM 1024
#define HD    64
#define NHEAD 16

typedef __bf16 bf16;
typedef __attribute__((ext_vector_type(16))) __bf16 v16bf;
typedef __attribute__((ext_vector_type(8)))  __bf16 v8bf;
typedef __attribute__((ext_vector_type(8)))  float  v8f;

static __device__ __forceinline__ v8f wmma_bf16(v16bf a, v16bf b, v8f c) {
  // (neg_a, A, neg_b, B, c_mod, C, reuse_a, reuse_b)
  return __builtin_amdgcn_wmma_f32_16x16x32_bf16(false, a, false, b, (short)0, c,
                                                 false, false);
}

// 16 contiguous halves (two aligned 16B LDS reads)
static __device__ __forceinline__ v16bf ld16c(const bf16* p) {
  v8bf lo = *(const v8bf*)p;
  v8bf hi = *(const v8bf*)(p + 8);
  return __builtin_shufflevector(lo, hi, 0,1,2,3,4,5,6,7,8,9,10,11,12,13,14,15);
}
// halves [0..7] and [16..23] (A-fragment gap layout, ISA 7.12.2)
static __device__ __forceinline__ v16bf ld16g(const bf16* p) {
  v8bf lo = *(const v8bf*)p;
  v8bf hi = *(const v8bf*)(p + 16);
  return __builtin_shufflevector(lo, hi, 0,1,2,3,4,5,6,7,8,9,10,11,12,13,14,15);
}

// ---------------------------------------------------------------------------
// CDNA5 async global->LDS copy (ASYNCcnt), 16B per enabled lane.
// VDST operand = wave-relative LDS byte address; low 32 bits of the generic
// pointer to a __shared__ object are exactly that (aperture: {SHARED_BASE,off}).
// ---------------------------------------------------------------------------
static __device__ __forceinline__ unsigned lds_off(const void* p) {
  return (unsigned)(unsigned long long)p;
}
static __device__ __forceinline__ void async_g2l_b128(unsigned lds_byte_off,
                                                      const void* gaddr) {
  asm volatile("global_load_async_to_lds_b128 %0, %1, off"
               :
               : "v"(lds_byte_off), "v"((unsigned long long)gaddr)
               : "memory");
}
static __device__ __forceinline__ void wait_async0() {
  asm volatile("s_wait_asynccnt 0x0" ::: "memory");
}

// ---------------------------------------------------------------------------
// fp32 -> bf16 conversion (grid-stride)
// ---------------------------------------------------------------------------
__global__ __launch_bounds__(256) void cvt_kernel(const float* __restrict__ in,
                                                  bf16* __restrict__ out, long n) {
  long i = (long)blockIdx.x * blockDim.x + threadIdx.x;
  long stride = (long)gridDim.x * blockDim.x;
  for (; i < n; i += stride) out[i] = (bf16)in[i];
}

// ---------------------------------------------------------------------------
// Generic bf16 WMMA GEMM: C[M,N] = A[M,K] @ B[K,N] (+bias) with fused epilogue
// Block tile 64x128x32, 256 threads = 8 waves, each wave 32x32 (2x2 WMMA).
// ---------------------------------------------------------------------------
struct Epi {
  int mode;             // 0:bf16  1:f32  2:res+f32  3:gelu->bf16
                        // 4:split gated +=   5:gated +=
  const float* bias;    // [N]
  bf16*  outB;
  float* outF;
  int    ldO;
  const float* res;     // mode 2, stride ldO
  float* dst1;          // modes 4/5 (+=, stride C_DIM)
  float* dst2;          // mode 4
  const float* g1;      // gates, row stride gStride
  const float* g2;
  int gStride;
  int halfRows;         // mode 4: rows per half per bt (=N tokens)
  int rowsPerBT;        // mode 5
};

__global__ __launch_bounds__(256) void gemm_bf16_kernel(
    const bf16* __restrict__ A, const bf16* __restrict__ B,
    int M, int N, int K, Epi e) {
  __shared__ bf16 As[64 * 40];    // [m][k], padded stride 40 halves
  __shared__ bf16 Bs[128 * 40];   // [n][k] (transposed at load)

  const int tid  = threadIdx.x;
  const int w    = tid >> 5, lane = tid & 31, l16 = lane & 15, hi = lane >> 4;
  const int wm   = w >> 2,  wn = w & 3;
  const int rowBase = blockIdx.y * 64, colBase = blockIdx.x * 128;

  v8f acc[2][2];
#pragma unroll
  for (int mi = 0; mi < 2; mi++)
#pragma unroll
    for (int ni = 0; ni < 2; ni++)
#pragma unroll
      for (int q = 0; q < 8; q++) acc[mi][ni][q] = 0.f;

  const int arow = tid >> 2, acol = (tid & 3) * 8;
  const unsigned aLds = lds_off(&As[arow * 40 + acol]);

  for (int k0 = 0; k0 < K; k0 += 32) {
    // --- stage A (64x32) via async global->LDS copy ---
    if (rowBase + arow < M) {
      const bf16* ap = A + (size_t)(rowBase + arow) * K + k0 + acol;
      async_g2l_b128(aLds, ap);
      if (k0 + 32 < K) __builtin_prefetch(ap + 32, 0, 1);   // global_prefetch_b8
    } else {
      v8bf z;
#pragma unroll
      for (int j = 0; j < 8; j++) z[j] = (bf16)0.f;
      *(v8bf*)(&As[arow * 40 + acol]) = z;
    }

    // --- stage B (32x128), store transposed [n][k] ---
#pragma unroll
    for (int i = tid; i < 512; i += 256) {
      int kr = i >> 4, nc = (i & 15) * 8;
      v8bf bv = *(const v8bf*)(B + (size_t)(k0 + kr) * N + colBase + nc);
#pragma unroll
      for (int j = 0; j < 8; j++) Bs[(nc + j) * 40 + kr] = bv[j];
    }
    wait_async0();
    __syncthreads();

    v16bf afr[2], bfr[2];
#pragma unroll
    for (int mi = 0; mi < 2; mi++)
      afr[mi] = ld16g(&As[(wm * 32 + mi * 16 + l16) * 40 + (hi ? 8 : 0)]);
#pragma unroll
    for (int ni = 0; ni < 2; ni++)
      bfr[ni] = ld16g(&Bs[(wn * 32 + ni * 16 + l16) * 40 + (hi ? 8 : 0)]);
#pragma unroll
    for (int mi = 0; mi < 2; mi++)
#pragma unroll
      for (int ni = 0; ni < 2; ni++)
        acc[mi][ni] = wmma_bf16(afr[mi], bfr[ni], acc[mi][ni]);
    __syncthreads();
  }

  // --- fused epilogue (C layout: lane=N, vgpr e -> M = hi*8+e) ---
#pragma unroll
  for (int mi = 0; mi < 2; mi++) {
#pragma unroll
    for (int ni = 0; ni < 2; ni++) {
      const int col  = colBase + wn * 32 + ni * 16 + l16;
      const float bv = e.bias ? e.bias[col] : 0.f;
      const int row0 = rowBase + wm * 32 + mi * 16 + hi * 8;
#pragma unroll
      for (int q = 0; q < 8; q++) {
        const int row = row0 + q;
        if (row >= M) continue;
        float val = acc[mi][ni][q] + bv;
        if (e.mode == 0) {
          e.outB[(size_t)row * e.ldO + col] = (bf16)val;
        } else if (e.mode == 1) {
          e.outF[(size_t)row * e.ldO + col] = val;
        } else if (e.mode == 2) {
          e.outF[(size_t)row * e.ldO + col] =
              e.res[(size_t)row * e.ldO + col] + val;
        } else if (e.mode == 3) {
          float g = 0.5f * val *
                    (1.f + tanhf(0.7978845608028654f *
                                 (val + 0.044715f * val * val * val)));
          e.outB[(size_t)row * e.ldO + col] = (bf16)g;
        } else if (e.mode == 4) {
          const int span = 2 * e.halfRows;
          const int bt = row / span, n = row - bt * span;
          if (n < e.halfRows) {
            float* p = e.dst1 + (size_t)(bt * e.halfRows + n) * C_DIM + col;
            *p += e.g1[(size_t)bt * e.gStride + col] * val;
          } else {
            float* p = e.dst2 +
                       (size_t)(bt * e.halfRows + (n - e.halfRows)) * C_DIM + col;
            *p += e.g2[(size_t)bt * e.gStride + col] * val;
          }
        } else {  // mode 5
          const int bt = row / e.rowsPerBT;
          float* p = e.dst1 + (size_t)row * C_DIM + col;
          *p += e.g1[(size_t)bt * e.gStride + col] * val;
        }
      }
    }
  }
}

// ---------------------------------------------------------------------------
// Fused flash attention (softmax(Q K^T * scale) V), head dim 64, WMMA.
// Block: 32 q-rows x one head x one batch item; keys streamed in 64-chunks.
// K chunk is staged with async global->LDS; V is transposed at load.
// ---------------------------------------------------------------------------
__global__ __launch_bounds__(256) void attn_kernel(
    const bf16* __restrict__ Q, const bf16* __restrict__ Kp,
    const bf16* __restrict__ Vp, int ldQ, int ldKV, int nQ, int nK,
    float scale, bf16* __restrict__ O, int ldO) {
  __shared__ bf16  Qs[32 * 72];   // [q][d]
  __shared__ bf16  Ks[64 * 72];   // [key][d]   (chunk)
  __shared__ bf16  Vt[64 * 72];   // [d][key]   (chunk, transposed)
  __shared__ float Ss[32 * 68];   // scores f32
  __shared__ bf16  Ps[32 * 72];   // probabilities bf16
  __shared__ float rowM[32], rowS[32], rowF[32];

  const int b = blockIdx.z, h = blockIdx.y, qt = blockIdx.x;
  const int tid = threadIdx.x;
  const int w = tid >> 5, lane = tid & 31, l16 = lane & 15, hi = lane >> 4;

  const bf16* Qg = Q + (size_t)(b * nQ + qt * 32) * ldQ + h * HD;
  const bf16* Kg = Kp + (size_t)b * nK * ldKV + h * HD;
  const bf16* Vg = Vp + (size_t)b * nK * ldKV + h * HD;

  // load Q tile (32x64)
  {
    const int r = tid >> 3, c = (tid & 7) * 8;
    *(v8bf*)(&Qs[r * 72 + c]) = *(const v8bf*)(Qg + (size_t)r * ldQ + c);
  }
  if (tid < 32) { rowM[tid] = -3.0e38f; rowS[tid] = 0.f; }

  v8f acc;
#pragma unroll
  for (int e = 0; e < 8; e++) acc[e] = 0.f;

  const int qr = (w & 1) * 16;          // PV: q-subtile
  const int cb = (w >> 1) * 16;         // PV: head-dim subtile
  const int smi = (w & 1) * 16;         // S:  q-subtile
  const int skt = (w >> 1) * 16;        // S:  key subtile

  for (int c0 = 0; c0 < nK; c0 += 64) {
    __syncthreads();
    // stage K (async, row-major) and V (transposed) chunk
#pragma unroll
    for (int i = tid; i < 512; i += 256) {
      const int kr = i >> 3, cc = (i & 7) * 8;
      async_g2l_b128(lds_off(&Ks[kr * 72 + cc]),
                     Kg + (size_t)(c0 + kr) * ldKV + cc);
      v8bf vv = *(const v8bf*)(Vg + (size_t)(c0 + kr) * ldKV + cc);
#pragma unroll
      for (int j = 0; j < 8; j++) Vt[(cc + j) * 72 + kr] = vv[j];
    }
    wait_async0();
    __syncthreads();

    // S = Q K^T  (each wave: one 16x16 tile, K-dim = hd = 64 -> 2 WMMA)
    {
      v8f s;
#pragma unroll
      for (int e = 0; e < 8; e++) s[e] = 0.f;
#pragma unroll
      for (int k0 = 0; k0 < 64; k0 += 32) {
        v16bf a  = ld16g(&Qs[(smi + l16) * 72 + k0 + (hi ? 8 : 0)]);
        v16bf bb = ld16g(&Ks[(skt + l16) * 72 + k0 + (hi ? 8 : 0)]);
        s = wmma_bf16(a, bb, s);
      }
#pragma unroll
      for (int e = 0; e < 8; e++)
        Ss[(smi + hi * 8 + e) * 68 + skt + l16] = s[e] * scale;
    }
    __syncthreads();

    // online softmax update (8 threads per q-row)
    {
      const int r = tid >> 3, sl = tid & 7;
      float cm = -3.0e38f;
#pragma unroll
      for (int j = sl; j < 64; j += 8) cm = fmaxf(cm, Ss[r * 68 + j]);
      cm = fmaxf(cm, __shfl_xor(cm, 1));
      cm = fmaxf(cm, __shfl_xor(cm, 2));
      cm = fmaxf(cm, __shfl_xor(cm, 4));
      const float mo = rowM[r];
      const float mn = fmaxf(mo, cm);
      float ps = 0.f;
#pragma unroll
      for (int j = sl; j < 64; j += 8) {
        float ev = __expf(Ss[r * 68 + j] - mn);
        Ps[r * 72 + j] = (bf16)ev;
        ps += ev;
      }
      ps += __shfl_xor(ps, 1);
      ps += __shfl_xor(ps, 2);
      ps += __shfl_xor(ps, 4);
      if (sl == 0) {
        const float f = __expf(mo - mn);
        rowF[r] = f;
        rowS[r] = rowS[r] * f + ps;
        rowM[r] = mn;
      }
    }
    __syncthreads();

    // rescale accumulator, then O += P V over this chunk
#pragma unroll
    for (int e = 0; e < 8; e++) acc[e] *= rowF[qr + hi * 8 + e];
#pragma unroll
    for (int k0 = 0; k0 < 64; k0 += 32) {
      v16bf a  = ld16g(&Ps[(qr + l16) * 72 + k0 + (hi ? 8 : 0)]);
      v16bf bb = ld16g(&Vt[(cb + l16) * 72 + k0 + (hi ? 8 : 0)]);
      acc = wmma_bf16(a, bb, acc);
    }
  }

  bf16* Og =
      O + (size_t)(b * nQ + qt * 32 + qr + hi * 8) * ldO + h * HD + cb + l16;
#pragma unroll
  for (int e = 0; e < 8; e++)
    Og[(size_t)e * ldO] = (bf16)(acc[e] / rowS[qr + hi * 8 + e]);
}

// ---------------------------------------------------------------------------
// LayerNorm + adaLN modulate: out_bf16 = ln(x) * (1+sc[bt]) + sh[bt]
// Row remapping supports writing into the concat-for-self-attn layout.
// ---------------------------------------------------------------------------
__global__ __launch_bounds__(256) void lnmod_kernel(
    const float* __restrict__ x, const float* __restrict__ sh,
    const float* __restrict__ sc, int mStride, bf16* __restrict__ out,
    int outRowStride, int halfOff, int rowsPerBT) {
  const int row = blockIdx.x, tid = threadIdx.x;
  const float* xr = x + (size_t)row * C_DIM;
  float v[4], s = 0.f, s2 = 0.f;
#pragma unroll
  for (int i = 0; i < 4; i++) {
    v[i] = xr[tid + i * 256];
    s += v[i];
    s2 += v[i] * v[i];
  }
#pragma unroll
  for (int o = 1; o < 32; o <<= 1) {
    s += __shfl_xor(s, o);
    s2 += __shfl_xor(s2, o);
  }
  __shared__ float ws[8], ws2[8];
  if ((tid & 31) == 0) { ws[tid >> 5] = s; ws2[tid >> 5] = s2; }
  __syncthreads();
  float ts = 0.f, ts2 = 0.f;
#pragma unroll
  for (int j = 0; j < 8; j++) { ts += ws[j]; ts2 += ws2[j]; }
  const float mu   = ts * (1.f / 1024.f);
  const float var  = ts2 * (1.f / 1024.f) - mu * mu;
  const float rstd = rsqrtf(var + 1e-6f);
  const int bt = row / rowsPerBT, n = row - bt * rowsPerBT;
  bf16* orow = out + (size_t)(bt * outRowStride + halfOff + n) * C_DIM;
#pragma unroll
  for (int i = 0; i < 4; i++) {
    const int c = tid + i * 256;
    const float scv = sc[(size_t)bt * mStride + c];
    const float shv = sh[(size_t)bt * mStride + c];
    orow[c] = (bf16)(((v[i] - mu) * rstd) * (1.f + scv) + shv);
  }
}

// ---------------------------------------------------------------------------
// s = silu(repeat(t,T) + cond)   -> bf16 [BT, C]
// ---------------------------------------------------------------------------
__global__ __launch_bounds__(256) void silu_kernel(const float* __restrict__ t,
                                                   const float* __restrict__ cond,
                                                   int T, bf16* __restrict__ out) {
  const int bt = blockIdx.x, tid = threadIdx.x;
  const int b = bt / T;
#pragma unroll
  for (int i = 0; i < 4; i++) {
    const int c = tid + i * 256;
    const float v = t[(size_t)b * C_DIM + c] + cond[(size_t)bt * C_DIM + c];
    out[(size_t)bt * C_DIM + c] = (bf16)(v / (1.f + __expf(-v)));
  }
}

// ---------------------------------------------------------------------------
// Host orchestration
// ---------------------------------------------------------------------------
extern "C" void kernel_launch(void* const* d_in, const int* in_sizes, int n_in,
                              void* d_out, int out_size, void* d_ws,
                              size_t ws_size, hipStream_t stream) {
  const float* x1_in = (const float*)d_in[0];
  const float* x2_in = (const float*)d_in[1];
  const float* t_in  = (const float*)d_in[2];
  const float* vf1   = (const float*)d_in[3];
  const float* vf2   = (const float*)d_in[4];
  const float* cond  = (const float*)d_in[5];
  const float *Wq1 = (const float*)d_in[6],  *bq1 = (const float*)d_in[7];
  const float *Wkv1 = (const float*)d_in[8], *bkv1 = (const float*)d_in[9];
  const float *Wp1 = (const float*)d_in[10], *bp1 = (const float*)d_in[11];
  const float *Wq2 = (const float*)d_in[12], *bq2 = (const float*)d_in[13];
  const float *Wkv2 = (const float*)d_in[14], *bkv2 = (const float*)d_in[15];
  const float *Wp2 = (const float*)d_in[16], *bp2 = (const float*)d_in[17];
  const float *Wqkv = (const float*)d_in[18], *bqkv = (const float*)d_in[19];
  const float *Wpa = (const float*)d_in[20], *bpa = (const float*)d_in[21];
  const float *Wada1 = (const float*)d_in[22], *bada1 = (const float*)d_in[23];
  const float *Wada2 = (const float*)d_in[24], *bada2 = (const float*)d_in[25];
  const float *Wf1a = (const float*)d_in[26], *bf1a = (const float*)d_in[27];
  const float *Wf2a = (const float*)d_in[28], *bf2a = (const float*)d_in[29];
  const float *Wf1b = (const float*)d_in[30], *bf1b = (const float*)d_in[31];
  const float *Wf2b = (const float*)d_in[32], *bf2b = (const float*)d_in[33];

  const int Bb   = in_sizes[2] / C_DIM;          // batch (2)
  const int TN   = in_sizes[0] / (Bb * C_DIM);   // tokens per batch (2048)
  const int BT   = in_sizes[5] / C_DIM;          // B*T (16)
  const int T    = BT / Bb;                      // 8
  const int N    = TN / T;                       // tokens per frame (256)
  const int Mctx = in_sizes[3] / (Bb * C_DIM);   // ctx length (256)
  const int Rx   = Bb * TN;                      // 4096
  const int Rctx = Bb * Mctx;                    // 512
  const int Rcat = BT * 2 * N;                   // 8192

  float* x1o = (float*)d_out;                    // x1 accumulator == output
  float* x2o = x1o + (size_t)Rx * C_DIM;         // x2 accumulator == output

  size_t off = 0;
  auto allocB = [&](size_t n) -> bf16* {
    bf16* p = (bf16*)((char*)d_ws + off);
    off = (off + n * sizeof(bf16) + 255) & ~(size_t)255;
    return p;
  };
  auto allocF = [&](size_t n) -> float* {
    float* p = (float*)((char*)d_ws + off);
    off = (off + n * sizeof(float) + 255) & ~(size_t)255;
    return p;
  };

  bf16* wq1b  = allocB((size_t)C_DIM * C_DIM);
  bf16* wkv1b = allocB((size_t)C_DIM * 2 * C_DIM);
  bf16* wp1b  = allocB((size_t)C_DIM * C_DIM);
  bf16* wq2b  = allocB((size_t)C_DIM * C_DIM);
  bf16* wkv2b = allocB((size_t)C_DIM * 2 * C_DIM);
  bf16* wp2b  = allocB((size_t)C_DIM * C_DIM);
  bf16* wqkvb = allocB((size_t)C_DIM * 3 * C_DIM);
  bf16* wpab  = allocB((size_t)C_DIM * C_DIM);
  bf16* wada1b = allocB((size_t)C_DIM * 6 * C_DIM);
  bf16* wada2b = allocB((size_t)C_DIM * 6 * C_DIM);
  bf16* wf1ab = allocB((size_t)C_DIM * 4 * C_DIM);
  bf16* wf2ab = allocB((size_t)4 * C_DIM * C_DIM);
  bf16* wf1bb = allocB((size_t)C_DIM * 4 * C_DIM);
  bf16* wf2bb = allocB((size_t)4 * C_DIM * C_DIM);
  bf16* x1b  = allocB((size_t)Rx * C_DIM);
  bf16* x2b  = allocB((size_t)Rx * C_DIM);
  bf16* vf1b = allocB((size_t)Rctx * C_DIM);
  bf16* vf2b = allocB((size_t)Rctx * C_DIM);
  bf16* qbuf = allocB((size_t)Rx * C_DIM);
  bf16* kvbuf = allocB((size_t)Rctx * 2 * C_DIM);
  bf16* obuf = allocB((size_t)Rx * C_DIM);
  bf16* sbuf = allocB((size_t)BT * C_DIM);
  float* m1 = allocF((size_t)BT * 6 * C_DIM);
  float* m2 = allocF((size_t)BT * 6 * C_DIM);
  bf16* xcat = allocB((size_t)Rcat * C_DIM);
  bf16* qkvb = allocB((size_t)Rcat * 3 * C_DIM);
  bf16* oat  = allocB((size_t)Rcat * C_DIM);
  bf16* hx   = allocB((size_t)Rx * C_DIM);
  bf16* h1   = allocB((size_t)Rx * 4 * C_DIM);

  auto cvt = [&](const float* s, bf16* d, size_t n) {
    cvt_kernel<<<dim3(512), 256, 0, stream>>>(s, d, (long)n);
  };
  cvt(Wq1, wq1b, (size_t)C_DIM * C_DIM);
  cvt(Wkv1, wkv1b, (size_t)C_DIM * 2 * C_DIM);
  cvt(Wp1, wp1b, (size_t)C_DIM * C_DIM);
  cvt(Wq2, wq2b, (size_t)C_DIM * C_DIM);
  cvt(Wkv2, wkv2b, (size_t)C_DIM * 2 * C_DIM);
  cvt(Wp2, wp2b, (size_t)C_DIM * C_DIM);
  cvt(Wqkv, wqkvb, (size_t)C_DIM * 3 * C_DIM);
  cvt(Wpa, wpab, (size_t)C_DIM * C_DIM);
  cvt(Wada1, wada1b, (size_t)C_DIM * 6 * C_DIM);
  cvt(Wada2, wada2b, (size_t)C_DIM * 6 * C_DIM);
  cvt(Wf1a, wf1ab, (size_t)C_DIM * 4 * C_DIM);
  cvt(Wf2a, wf2ab, (size_t)4 * C_DIM * C_DIM);
  cvt(Wf1b, wf1bb, (size_t)C_DIM * 4 * C_DIM);
  cvt(Wf2b, wf2bb, (size_t)4 * C_DIM * C_DIM);
  cvt(x1_in, x1b, (size_t)Rx * C_DIM);
  cvt(x2_in, x2b, (size_t)Rx * C_DIM);
  cvt(vf1, vf1b, (size_t)Rctx * C_DIM);
  cvt(vf2, vf2b, (size_t)Rctx * C_DIM);

  auto gemm = [&](const bf16* A, const bf16* Bw, int M, int Nn, int K,
                  const Epi& e) {
    gemm_bf16_kernel<<<dim3(Nn / 128, (M + 63) / 64), 256, 0, stream>>>(
        A, Bw, M, Nn, K, e);
  };

  const float scale = 0.125f;  // hd^-0.5, hd = 64

  // ---- cross attention 1:  x1 = x1_in + proj(attn(q(x1), kv(v_fea))) ----
  { Epi e{}; e.mode = 0; e.bias = bq1; e.outB = qbuf; e.ldO = C_DIM;
    gemm(x1b, wq1b, Rx, C_DIM, C_DIM, e); }
  { Epi e{}; e.mode = 0; e.bias = bkv1; e.outB = kvbuf; e.ldO = 2 * C_DIM;
    gemm(vf1b, wkv1b, Rctx, 2 * C_DIM, C_DIM, e); }
  attn_kernel<<<dim3(TN / 32, NHEAD, Bb), 256, 0, stream>>>(
      qbuf, kvbuf, kvbuf + C_DIM, C_DIM, 2 * C_DIM, TN, Mctx, scale, obuf,
      C_DIM);
  { Epi e{}; e.mode = 2; e.bias = bp1; e.outF = x1o; e.ldO = C_DIM; e.res = x1_in;
    gemm(obuf, wp1b, Rx, C_DIM, C_DIM, e); }

  // ---- cross attention 2 ----
  { Epi e{}; e.mode = 0; e.bias = bq2; e.outB = qbuf; e.ldO = C_DIM;
    gemm(x2b, wq2b, Rx, C_DIM, C_DIM, e); }
  { Epi e{}; e.mode = 0; e.bias = bkv2; e.outB = kvbuf; e.ldO = 2 * C_DIM;
    gemm(vf2b, wkv2b, Rctx, 2 * C_DIM, C_DIM, e); }
  attn_kernel<<<dim3(TN / 32, NHEAD, Bb), 256, 0, stream>>>(
      qbuf, kvbuf, kvbuf + C_DIM, C_DIM, 2 * C_DIM, TN, Mctx, scale, obuf,
      C_DIM);
  { Epi e{}; e.mode = 2; e.bias = bp2; e.outF = x2o; e.ldO = C_DIM; e.res = x2_in;
    gemm(obuf, wp2b, Rx, C_DIM, C_DIM, e); }

  // ---- adaLN modulations ----
  silu_kernel<<<BT, 256, 0, stream>>>(t_in, cond, T, sbuf);
  { Epi e{}; e.mode = 1; e.bias = bada1; e.outF = m1; e.ldO = 6 * C_DIM;
    gemm(sbuf, wada1b, BT, 6 * C_DIM, C_DIM, e); }
  { Epi e{}; e.mode = 1; e.bias = bada2; e.outF = m2; e.ldO = 6 * C_DIM;
    gemm(sbuf, wada2b, BT, 6 * C_DIM, C_DIM, e); }

  // ---- LN + modulate into concatenated self-attn input ----
  lnmod_kernel<<<Rx, 256, 0, stream>>>(x1o, m1, m1 + C_DIM, 6 * C_DIM, xcat,
                                       2 * N, 0, N);
  lnmod_kernel<<<Rx, 256, 0, stream>>>(x2o, m2, m2 + C_DIM, 6 * C_DIM, xcat,
                                       2 * N, N, N);

  // ---- self attention ----
  { Epi e{}; e.mode = 0; e.bias = bqkv; e.outB = qkvb; e.ldO = 3 * C_DIM;
    gemm(xcat, wqkvb, Rcat, 3 * C_DIM, C_DIM, e); }
  attn_kernel<<<dim3((2 * N) / 32, NHEAD, BT), 256, 0, stream>>>(
      qkvb, qkvb + C_DIM, qkvb + 2 * C_DIM, 3 * C_DIM, 3 * C_DIM, 2 * N, 2 * N,
      scale, oat, C_DIM);
  { Epi e{}; e.mode = 4; e.bias = bpa; e.dst1 = x1o; e.dst2 = x2o;
    e.g1 = m1 + 2 * C_DIM; e.g2 = m2 + 2 * C_DIM; e.gStride = 6 * C_DIM;
    e.halfRows = N;
    gemm(oat, wpab, Rcat, C_DIM, C_DIM, e); }

  // ---- MLP 1 (x1) ----
  lnmod_kernel<<<Rx, 256, 0, stream>>>(x1o, m1 + 3 * C_DIM, m1 + 4 * C_DIM,
                                       6 * C_DIM, hx, N, 0, N);
  { Epi e{}; e.mode = 3; e.bias = bf1a; e.outB = h1; e.ldO = 4 * C_DIM;
    gemm(hx, wf1ab, Rx, 4 * C_DIM, C_DIM, e); }
  { Epi e{}; e.mode = 5; e.bias = bf2a; e.dst1 = x1o; e.g1 = m1 + 5 * C_DIM;
    e.gStride = 6 * C_DIM; e.rowsPerBT = N;
    gemm(h1, wf2ab, Rx, C_DIM, 4 * C_DIM, e); }

  // ---- MLP 2 (x2) ----
  lnmod_kernel<<<Rx, 256, 0, stream>>>(x2o, m2 + 3 * C_DIM, m2 + 4 * C_DIM,
                                       6 * C_DIM, hx, N, 0, N);
  { Epi e{}; e.mode = 3; e.bias = bf1b; e.outB = h1; e.ldO = 4 * C_DIM;
    gemm(hx, wf1bb, Rx, 4 * C_DIM, C_DIM, e); }
  { Epi e{}; e.mode = 5; e.bias = bf2b; e.dst1 = x2o; e.g1 = m2 + 5 * C_DIM;
    e.gStride = 6 * C_DIM; e.rowsPerBT = N;
    gemm(h1, wf2bb, Rx, C_DIM, 4 * C_DIM, e); }

  (void)n_in; (void)out_size; (void)ws_size;
}
